// FluxSingleTransformerBlock_36112085025541
// MI455X (gfx1250) — compile-verified, matrix operand order checked by reading, not run on hardware
//
#include <hip/hip_runtime.h>
#include <hip/hip_bf16.h>
#include <math.h>

// ---------------- problem constants ----------------
constexpr int Tt  = 2048;
constexpr int Dd  = 3072;
constexpr int Hh  = 24;
constexpr int DH  = 128;
constexpr int MLP = 4 * Dd;
constexpr float EPSf = 1e-6f;

typedef __bf16 bf16_t;
typedef __attribute__((ext_vector_type(16))) __bf16 v16bf;
typedef __attribute__((ext_vector_type(8)))  __bf16 v8bf;
typedef __attribute__((ext_vector_type(8)))  float  v8f;
typedef __attribute__((ext_vector_type(4)))  int    v4i;

// Async global->LDS copy (CDNA5 GLOBAL_LOAD_ASYNC_TO_LDS_B128, ASYNCcnt)
#if defined(__has_builtin)
#  if __has_builtin(__builtin_amdgcn_global_load_async_to_lds_b128) && \
      __has_builtin(__builtin_amdgcn_s_wait_asynccnt)
#    define USE_ASYNC_LDS 1
#  endif
#endif

__device__ inline void copy16(const bf16_t* g, bf16_t* l) {
#ifdef USE_ASYNC_LDS
    auto l3 = (__attribute__((address_space(3))) v4i*)(unsigned)(size_t)l;
    auto g1 = (__attribute__((address_space(1))) v4i*)(size_t)g;
    __builtin_amdgcn_global_load_async_to_lds_b128(g1, l3, 0, 0);
#else
    *reinterpret_cast<v8bf*>(l) = *reinterpret_cast<const v8bf*>(g);
#endif
}

__device__ inline void wait_async_stage() {
#ifdef USE_ASYNC_LDS
    __builtin_amdgcn_s_wait_asynccnt(0);
#endif
}

// WMMA: D = A(16x32 bf16) * B(32x16 bf16) + C(16x16 f32)
__device__ inline v8f wmma_bf16(v16bf a, v16bf b, v8f c) {
    return __builtin_amdgcn_wmma_f32_16x16x32_bf16(
        false, a, false, b, (short)0, c, false, false);
}

// Lane fragment: two 16B chunks per cdna5 16-bit A/B layout
// (lanes 0-15 -> K base 0, lanes 16-31 -> K base 8).
__device__ inline v16bf load_frag(const bf16_t* rowptr, int kb) {
    v8bf c0 = *reinterpret_cast<const v8bf*>(rowptr + kb);
    v8bf c1 = *reinterpret_cast<const v8bf*>(rowptr + 16 + kb);
    v16bf r;
#pragma unroll
    for (int i = 0; i < 8; ++i) { r[i] = c0[i]; r[i + 8] = c1[i]; }
    return r;
}

__device__ inline float gelu_tanh(float x) {
    float x3 = x * x * x;
    return 0.5f * x * (1.0f + tanhf(0.7978845608028654f * (x + 0.044715f * x3)));
}

// ---------------- 1. adaLN embedding --------------------------------------
__global__ void ada_kernel(const float* __restrict__ temb,
                           const float* __restrict__ ada_w,
                           const float* __restrict__ ada_b,
                           float* __restrict__ emb) {
    __shared__ float st[Dd];
    for (int i = threadIdx.x; i < Dd; i += blockDim.x) {
        float tv = temb[i];
        st[i] = tv * (1.0f / (1.0f + __expf(-tv)));
    }
    __syncthreads();
    int j = blockIdx.x * blockDim.x + threadIdx.x;
    if (j >= 3 * Dd) return;
    float acc = ada_b[j];
    for (int i = 0; i < Dd; ++i)
        acc = fmaf(st[i], ada_w[(size_t)i * (3 * Dd) + j], acc);
    emb[j] = acc;
}

// ---------------- 2. LayerNorm + modulation (bf16 out) --------------------
__global__ __launch_bounds__(256)
void ln_mod_kernel(const float* __restrict__ hs,
                   const float* __restrict__ emb,
                   bf16_t* __restrict__ x) {
    int t = blockIdx.x, tid = threadIdx.x;
    const float* row = hs + (size_t)t * Dd;
    float s = 0.f, s2 = 0.f;
    for (int d = tid; d < Dd; d += 256) { float v = row[d]; s += v; s2 += v * v; }
#pragma unroll
    for (int off = 16; off >= 1; off >>= 1) {
        s  += __shfl_xor(s,  off, 32);
        s2 += __shfl_xor(s2, off, 32);
    }
    __shared__ float w1[8], w2[8];
    if ((tid & 31) == 0) { w1[tid >> 5] = s; w2[tid >> 5] = s2; }
    __syncthreads();
    s = 0.f; s2 = 0.f;
#pragma unroll
    for (int i = 0; i < 8; ++i) { s += w1[i]; s2 += w2[i]; }
    float mu  = s * (1.0f / Dd);
    float var = s2 * (1.0f / Dd) - mu * mu;
    float inv = rsqrtf(var + EPSf);
    bf16_t* xo = x + (size_t)t * Dd;
    for (int d = tid; d < Dd; d += 256) {
        float v = (row[d] - mu) * inv;
        xo[d] = (bf16_t)(v * emb[3 * d + 1] + emb[3 * d + 0]);
    }
}

// ---------------- 3. bf16 WMMA GEMM ---------------------------------------
// A: [M,K] bf16 (async-copied to LDS), W: [K,N] f32 (converted on stage).
// Block tile 128x128, K-step 64, 8 waves (4x2), wave tile 32x64,
// 16 WMMAs per barrier pair.
template <int ACT, int OBF>
__global__ __launch_bounds__(256)
void gemm_bf16_kernel(const bf16_t* __restrict__ A,
                      const float* __restrict__ W,
                      const float* __restrict__ bias,
                      void* __restrict__ Cout,
                      int M, int N, int K) {
    __shared__ bf16_t lds_a[128][72];   // rows 144B: 16B-aligned chunks
    __shared__ bf16_t lds_b[128][72];   // stored [N][K]
    const int tid  = threadIdx.x;
    const int lane = tid & 31, wid = tid >> 5;
    const int wave_m = wid & 3, wave_n = wid >> 2;       // 4 x 2 wave grid
    const int m0 = blockIdx.x * 128, n0 = blockIdx.y * 128;
    const int mbw = wave_m * 32, nbw = wave_n * 64;
    const int ln = lane & 15, kb = (lane >> 4) * 8;

    v8f acc[2][4];
#pragma unroll
    for (int i = 0; i < 2; ++i)
#pragma unroll
        for (int j = 0; j < 4; ++j) acc[i][j] = (v8f){};

    for (int k0 = 0; k0 < K; k0 += 64) {
        // A tile 128x64 bf16: 1024 x 16B chunks, 4 per thread (async)
#pragma unroll
        for (int r = 0; r < 4; ++r) {
            int idx = r * 256 + tid;
            int m = idx >> 3, c = idx & 7;
            copy16(A + (size_t)(m0 + m) * K + k0 + c * 8, &lds_a[m][c * 8]);
        }
        // W tile 64x128 f32 -> lds_b[n][k], packed bf16 pairs (one b32 store)
#pragma unroll
        for (int r = 0; r < 16; ++r) {
            int idx = r * 256 + tid;            // pair index 0..4095
            int n = idx & 127, kp = idx >> 7;   // kp: 0..31
            float w0 = W[(size_t)(k0 + 2 * kp) * N + n0 + n];
            float w1 = W[(size_t)(k0 + 2 * kp + 1) * N + n0 + n];
            union { bf16_t h[2]; unsigned u; } p;
            p.h[0] = (bf16_t)w0; p.h[1] = (bf16_t)w1;
            *reinterpret_cast<unsigned*>(&lds_b[n][2 * kp]) = p.u;
        }
        wait_async_stage();
        __syncthreads();
        // prefetch next K-slab while compute runs
        if (k0 + 64 < K) {
            __builtin_prefetch(W + (size_t)(k0 + 64 + (tid >> 2)) * N + n0 + (tid & 3) * 32, 0, 1);
            __builtin_prefetch(A + (size_t)(m0 + (tid >> 1)) * K + k0 + 64 + (tid & 1) * 32, 0, 1);
        }
#pragma unroll
        for (int kc = 0; kc < 2; ++kc) {
            v16bf af[2], bf[4];
#pragma unroll
            for (int i = 0; i < 2; ++i)
                af[i] = load_frag(&lds_a[mbw + i * 16 + ln][kc * 32], kb);
#pragma unroll
            for (int j = 0; j < 4; ++j)
                bf[j] = load_frag(&lds_b[nbw + j * 16 + ln][kc * 32], kb);
#pragma unroll
            for (int i = 0; i < 2; ++i)
#pragma unroll
                for (int j = 0; j < 4; ++j)
                    acc[i][j] = wmma_bf16(af[i], bf[j], acc[i][j]);
        }
        __syncthreads();
    }
    // epilogue: VGPR e -> row e (lanes 0-15) / e+8 (lanes 16-31)
    const int half8 = (lane >> 4) * 8;
#pragma unroll
    for (int i = 0; i < 2; ++i) {
#pragma unroll
        for (int j = 0; j < 4; ++j) {
            int ncol = n0 + nbw + j * 16 + ln;
            float bv = bias[ncol];
#pragma unroll
            for (int e = 0; e < 8; ++e) {
                int mrow = m0 + mbw + i * 16 + half8 + e;
                float v = acc[i][j][e] + bv;
                if (ACT == 1) v = gelu_tanh(v);
                if (OBF == 1)
                    ((bf16_t*)Cout)[(size_t)mrow * N + ncol] = (bf16_t)v;
                else
                    ((float*)Cout)[(size_t)mrow * N + ncol] = v;
            }
        }
    }
}

// ---------------- 4. q/k RMSNorm + RoPE, bf16 repack ----------------------
__global__ __launch_bounds__(64)
void qkv_post_kernel(const float* __restrict__ qkv,
                     const float* __restrict__ cosb,
                     const float* __restrict__ sinb,
                     const float* __restrict__ nqw,
                     const float* __restrict__ nkw,
                     bf16_t* __restrict__ qw, bf16_t* __restrict__ kw,
                     bf16_t* __restrict__ vt) {
    int t = blockIdx.x / Hh, h = blockIdx.x % Hh;
    int i = threadIdx.x;                       // pair (2i, 2i+1)
    const float* base = qkv + (size_t)t * (3 * Dd);
    float q0 = base[h * DH + 2 * i],          q1 = base[h * DH + 2 * i + 1];
    float k0 = base[Dd + h * DH + 2 * i],     k1 = base[Dd + h * DH + 2 * i + 1];
    float v0 = base[2 * Dd + h * DH + 2 * i], v1 = base[2 * Dd + h * DH + 2 * i + 1];
    float sq = q0 * q0 + q1 * q1, sk = k0 * k0 + k1 * k1;
#pragma unroll
    for (int off = 1; off < 32; off <<= 1) {
        sq += __shfl_xor(sq, off, 32);
        sk += __shfl_xor(sk, off, 32);
    }
    __shared__ float pq[2], pk[2];
    if ((i & 31) == 0) { pq[i >> 5] = sq; pk[i >> 5] = sk; }
    __syncthreads();
    sq = pq[0] + pq[1]; sk = pk[0] + pk[1];
    float rq = rsqrtf(sq * (1.0f / DH) + EPSf);
    float rk = rsqrtf(sk * (1.0f / DH) + EPSf);
    q0 *= rq * nqw[2 * i]; q1 *= rq * nqw[2 * i + 1];
    k0 *= rk * nkw[2 * i]; k1 *= rk * nkw[2 * i + 1];
    float c = cosb[t * (DH / 2) + i], s = sinb[t * (DH / 2) + i];
    float oq0 = q0 * c - q1 * s, oq1 = q0 * s + q1 * c;
    float ok0 = k0 * c - k1 * s, ok1 = k0 * s + k1 * c;
    size_t qo = ((size_t)h * Tt + t) * DH + 2 * i;
    qw[qo] = (bf16_t)oq0; qw[qo + 1] = (bf16_t)oq1;
    kw[qo] = (bf16_t)ok0; kw[qo + 1] = (bf16_t)ok1;
    size_t vo = ((size_t)h * DH + 2 * i) * Tt + t;
    vt[vo] = (bf16_t)v0; vt[vo + Tt] = (bf16_t)v1;
}

// ---------------- 5. flash attention (bf16 WMMA, bf16 out) ----------------
__global__ __launch_bounds__(128)
void attn_kernel(const bf16_t* __restrict__ qw,
                 const bf16_t* __restrict__ kw,
                 const bf16_t* __restrict__ vt,
                 bf16_t* __restrict__ attn) {
    __shared__ bf16_t p_lds[4][16][40];
    const int wid = threadIdx.x >> 5, lane = threadIdx.x & 31;
    const int h = blockIdx.x % Hh;
    const int q0 = (blockIdx.x / Hh) * 64 + wid * 16;
    const int kb = (lane >> 4) * 8;
    const int ln = lane & 15;
    const float sc = 0.08838834764831845f;   // 1/sqrt(128)

    const bf16_t* qrow = qw + ((size_t)h * Tt + q0 + ln) * DH;
    v16bf qf[4];
#pragma unroll
    for (int c = 0; c < 4; ++c) qf[c] = load_frag(qrow + c * 32, kb);

    v8f o[8]; float m[8], l[8];
#pragma unroll
    for (int j = 0; j < 8; ++j) o[j] = (v8f){};
#pragma unroll
    for (int e = 0; e < 8; ++e) { m[e] = -3.0e38f; l[e] = 0.f; }

    for (int kv = 0; kv < Tt; kv += 32) {
        v8f sacc[2] = {{}, {}};
#pragma unroll
        for (int j = 0; j < 2; ++j) {
            const bf16_t* krow = kw + ((size_t)h * Tt + kv + j * 16 + ln) * DH;
#pragma unroll
            for (int c = 0; c < 4; ++c) {
                v16bf kf = load_frag(krow + c * 32, kb);
                sacc[j] = wmma_bf16(qf[c], kf, sacc[j]);
            }
        }
#pragma unroll
        for (int e = 0; e < 8; ++e) {
            float a = sacc[0][e] * sc, b = sacc[1][e] * sc;
            float mx = fmaxf(a, b);
#pragma unroll
            for (int off = 1; off < 16; off <<= 1)
                mx = fmaxf(mx, __shfl_xor(mx, off, 32));
            float mn = fmaxf(m[e], mx);
            float corr = __expf(m[e] - mn);
            float p0 = __expf(a - mn), p1 = __expf(b - mn);
            float rs = p0 + p1;
#pragma unroll
            for (int off = 1; off < 16; off <<= 1)
                rs += __shfl_xor(rs, off, 32);
            l[e] = l[e] * corr + rs;
            m[e] = mn;
#pragma unroll
            for (int j8 = 0; j8 < 8; ++j8) o[j8][e] *= corr;
            int prow = e + ((lane >> 4) << 3);
            p_lds[wid][prow][ln]      = (bf16_t)p0;
            p_lds[wid][prow][ln + 16] = (bf16_t)p1;
        }
        __syncthreads();
        v16bf pf = load_frag(&p_lds[wid][ln][0], kb);
        __syncthreads();
#pragma unroll
        for (int j8 = 0; j8 < 8; ++j8) {
            const bf16_t* vrow = vt + ((size_t)h * DH + j8 * 16 + ln) * Tt + kv;
            v16bf vf = load_frag(vrow, kb);
            o[j8] = wmma_bf16(pf, vf, o[j8]);
        }
    }
    const int half8 = (lane >> 4) * 8;
#pragma unroll
    for (int j8 = 0; j8 < 8; ++j8) {
#pragma unroll
        for (int e = 0; e < 8; ++e) {
            int t = q0 + half8 + e;
            int dh = j8 * 16 + ln;
            attn[(size_t)t * Dd + h * DH + dh] = (bf16_t)(o[j8][e] / l[e]);
        }
    }
}

// ---------------- 6. residual + gate --------------------------------------
__global__ void final_kernel(const float* __restrict__ resid,
                             const float* __restrict__ emb,
                             const float* __restrict__ attn_out,
                             const float* __restrict__ mlp,
                             float* __restrict__ out) {
    size_t idx = (size_t)blockIdx.x * 256 + threadIdx.x;
    int d = (int)(idx % Dd);
    out[idx] = resid[idx] + emb[3 * d + 2] * (attn_out[idx] + mlp[idx]);
}

// ---------------- host launcher -------------------------------------------
extern "C" void kernel_launch(void* const* d_in, const int* in_sizes, int n_in,
                              void* d_out, int out_size, void* d_ws, size_t ws_size,
                              hipStream_t stream) {
    (void)in_sizes; (void)n_in; (void)out_size; (void)ws_size;
    const float* hs     = (const float*)d_in[0];
    const float* temb   = (const float*)d_in[1];
    const float* rcos   = (const float*)d_in[2];
    const float* rsin   = (const float*)d_in[3];
    const float* ada_w  = (const float*)d_in[4];
    const float* ada_b  = (const float*)d_in[5];
    const float* w_qkv  = (const float*)d_in[6];
    const float* b_qkv  = (const float*)d_in[7];
    const float* nqw    = (const float*)d_in[8];
    const float* nkw    = (const float*)d_in[9];
    const float* w_out  = (const float*)d_in[10];
    const float* b_out  = (const float*)d_in[11];
    const float* w_fc1  = (const float*)d_in[12];
    const float* b_fc1  = (const float*)d_in[13];
    const float* w_fc2  = (const float*)d_in[14];
    const float* b_fc2  = (const float*)d_in[15];
    float* out = (float*)d_out;

    char* w = (char*)d_ws;
    float*  emb      = (float*) (w + 0);            //  9216 f32
    bf16_t* x        = (bf16_t*)(w + 36864);        //  T*D bf16
    float*  qkv      = (float*) (w + 12619776);     //  T*3D f32
    bf16_t* qw       = (bf16_t*)(w + 88117248);     //  H*T*DH bf16
    bf16_t* kw       = (bf16_t*)(w + 100700160);
    bf16_t* vt       = (bf16_t*)(w + 113283072);
    bf16_t* attn     = (bf16_t*)(w + 125865984);    //  T*D bf16
    float*  attn_out = (float*) (w + 138448896);    //  T*D f32
    bf16_t* h1       = (bf16_t*)(w + 163614720);    //  T*MLP bf16
    float*  mlp      = (float*) (w + 213946368);    //  T*D f32

    ada_kernel<<<(3 * Dd + 255) / 256, 256, 0, stream>>>(temb, ada_w, ada_b, emb);
    ln_mod_kernel<<<Tt, 256, 0, stream>>>(hs, emb, x);
    // qkv GEMM  (2048 x 3072) @ (3072 x 9216)
    gemm_bf16_kernel<0, 0><<<dim3(Tt / 128, (3 * Dd) / 128), 256, 0, stream>>>(
        x, w_qkv, b_qkv, qkv, Tt, 3 * Dd, Dd);
    qkv_post_kernel<<<Tt * Hh, 64, 0, stream>>>(qkv, rcos, rsin, nqw, nkw, qw, kw, vt);
    attn_kernel<<<(Tt / 64) * Hh, 128, 0, stream>>>(qw, kw, vt, attn);
    // out projection
    gemm_bf16_kernel<0, 0><<<dim3(Tt / 128, Dd / 128), 256, 0, stream>>>(
        attn, w_out, b_out, attn_out, Tt, Dd, Dd);
    // fc1 + gelu -> bf16
    gemm_bf16_kernel<1, 1><<<dim3(Tt / 128, MLP / 128), 256, 0, stream>>>(
        x, w_fc1, b_fc1, h1, Tt, MLP, Dd);
    // fc2
    gemm_bf16_kernel<0, 0><<<dim3(Tt / 128, Dd / 128), 256, 0, stream>>>(
        h1, w_fc2, b_fc2, mlp, Tt, Dd, MLP);
    final_kernel<<<(Tt * Dd) / 256, 256, 0, stream>>>(hs, emb, attn_out, mlp, out);
}
